// SimpleGCN_21225728377317
// MI455X (gfx1250) — matrix operand order repaired and checked by reading, compiled, not run on hardware
//
#include <hip/hip_runtime.h>
#include <hip/hip_bf16.h>

// ---------------------------------------------------------------------------
// 2-layer GCN for MI455X (gfx1250, wave32, WMMA).
//   norm = dinv[src]*dinv[dst]  ->  fold into GEMM epilogue (pre-scale) and
//   aggregation epilogue (post-scale); self-loop folded into epilogue too.
// GEMMs use v_wmma_f32_16x16x32_f16 (f16 in, f32 accumulate).
// ---------------------------------------------------------------------------

typedef __attribute__((ext_vector_type(16))) _Float16 v16h;
typedef __attribute__((ext_vector_type(8)))  float    v8f;

#define N_NODES 50000
#define N_EDGES 400000

// ---------------- degree / dinv ----------------
__global__ void deg_count(const long long* __restrict__ ei, float* __restrict__ deg, int E) {
    int e = blockIdx.x * blockDim.x + threadIdx.x;
    if (e < E) atomicAdd(&deg[ei[E + e]], 1.0f);   // dst row of edge_index
}

__global__ void dinv_kernel(const float* __restrict__ deg, float* __restrict__ dinv, int n) {
    int i = blockIdx.x * blockDim.x + threadIdx.x;
    if (i < n) dinv[i] = rsqrtf(deg[i] + 1.0f);    // +1 self-loop; always > 0
}

// ---------------- W pre-swizzle into B-fragment layout ----------------
// B fragment as this code orders k (16-bit, 32x16 KxN per tile, same order as A), wave32:
//   lane L holds column N = (L&15); halves j=0..15 hold K = j + 16*(L>>4).
// Swizzled buffer: blocks of 512 halves per (ktile,ntile), half index =
//   blk*512 + lane*16 + j  -> each lane's fragment is one contiguous v16h.
__global__ void swizzle_w(const float* __restrict__ W, _Float16* __restrict__ Wsw,
                          int KT, int NT, int Nreal) {
    int gid = blockIdx.x * blockDim.x + threadIdx.x;
    int total = KT * NT * 512;
    if (gid >= total) return;
    int j    = gid & 15;
    int lane = (gid >> 4) & 31;
    int blk  = gid >> 9;
    int nt   = blk % NT;
    int kt   = blk / NT;
    int K = kt * 32 + j + 16 * (lane >> 4);
    int n = nt * 16 + (lane & 15);
    float v = (n < Nreal) ? W[(size_t)K * Nreal + n] : 0.0f;
    Wsw[gid] = (_Float16)v;
}

// ---------------- WMMA GEMM with dinv pre-scale epilogue ----------------
// Out[m, ng*64 + t*16 + n] = dinv[m] * sum_k A[m,k]*W[k,n]
// One wave per 16x64 output tile; A is f32 row-major (lda = K), converted
// to f16 in-register; B comes pre-swizzled (one v16h per lane per tile).
__global__ __launch_bounds__(32)
void gcn_wmma_gemm(const float* __restrict__ A, const _Float16* __restrict__ Wsw,
                   const float* __restrict__ dinv, float* __restrict__ Out,
                   int K, int NT, int ldo) {
    const int mt   = blockIdx.x;
    const int ng   = blockIdx.y;           // group of 4 n-tiles
    const int lane = threadIdx.x;          // 0..31
    const int row  = lane & 15;
    const int hi   = lane >> 4;

    const float* arow = A + (size_t)(mt * 16 + row) * K;
    const v16h*  bbas = (const v16h*)Wsw;  // 1 v16h = 16 halves; 32 per block

    v8f c0 = {}, c1 = {}, c2 = {}, c3 = {};
    const int nt0 = ng * 4;

    for (int k = 0; k < K; k += 32) {
        // A fragment (16-bit A 16x32), this code's k order (same as B): lane L (M=L&15) halves j:
        //   K = kb + j, kb = k + 16*(L>>4)
        const int kb = k + hi * 16;
        float4 f0 = *(const float4*)(arow + kb);
        float4 f1 = *(const float4*)(arow + kb + 4);
        float4 f2 = *(const float4*)(arow + kb + 8);
        float4 f3 = *(const float4*)(arow + kb + 12);
        v16h a;
        a[0]  = (_Float16)f0.x; a[1]  = (_Float16)f0.y;
        a[2]  = (_Float16)f0.z; a[3]  = (_Float16)f0.w;
        a[4]  = (_Float16)f1.x; a[5]  = (_Float16)f1.y;
        a[6]  = (_Float16)f1.z; a[7]  = (_Float16)f1.w;
        a[8]  = (_Float16)f2.x; a[9]  = (_Float16)f2.y;
        a[10] = (_Float16)f2.z; a[11] = (_Float16)f2.w;
        a[12] = (_Float16)f3.x; a[13] = (_Float16)f3.y;
        a[14] = (_Float16)f3.z; a[15] = (_Float16)f3.w;

        const int kt = k >> 5;
        const v16h* bb = bbas + (size_t)(kt * NT + nt0) * 32 + lane;
        v16h b0 = bb[0];
        v16h b1 = bb[32];
        v16h b2 = bb[64];
        v16h b3 = bb[96];

        c0 = __builtin_amdgcn_wmma_f32_16x16x32_f16(false, a, false, b0, (short)0, c0, false, false);
        c1 = __builtin_amdgcn_wmma_f32_16x16x32_f16(false, a, false, b1, (short)0, c1, false, false);
        c2 = __builtin_amdgcn_wmma_f32_16x16x32_f16(false, a, false, b2, (short)0, c2, false, false);
        c3 = __builtin_amdgcn_wmma_f32_16x16x32_f16(false, a, false, b3, (short)0, c3, false, false);
    }

    // C/D layout: VGPR r of lane L -> M = r + 8*(L>>4), N = L&15.
    #pragma unroll
    for (int r = 0; r < 8; ++r) {
        const int mo = mt * 16 + hi * 8 + r;
        const float dv = dinv[mo];
        float* orow = Out + (size_t)mo * ldo + ng * 64 + row;
        orow[0]  = dv * c0[r];
        orow[16] = dv * c1[r];
        orow[32] = dv * c2[r];
        orow[48] = dv * c3[r];
    }
}

// ---------------- edge scatter-add: agg[dst] += hp[src] ----------------
// One block per edge; threads cover a row in float4 chunks. Rows are hot in
// the 192MB L2, so the f32 atomics land in-cache.
__global__ void scatter_add(const long long* __restrict__ ei,
                            const float* __restrict__ hp, float* __restrict__ agg,
                            int C, int E) {
    const int e = blockIdx.x;
    const long long s = ei[e];
    const long long d = ei[E + e];
    const float4* srow = (const float4*)(hp + (size_t)s * C);
    float* drow = agg + (size_t)d * C;
    const int t = threadIdx.x;
    float4 v = srow[t];
    atomicAdd(drow + 4 * t + 0, v.x);
    atomicAdd(drow + 4 * t + 1, v.y);
    atomicAdd(drow + 4 * t + 2, v.z);
    atomicAdd(drow + 4 * t + 3, v.w);
}

// ---------------- epilogue: out = relu((agg + hp_self)*dinv + b) ----------
__global__ void post_kernel(const float* __restrict__ agg, const float* __restrict__ hp,
                            const float* __restrict__ dinv, const float* __restrict__ b,
                            float* __restrict__ out, int strideIn, int Cout, int strideOut) {
    const int rowi = blockIdx.x;
    const float dv = dinv[rowi];
    const float* ar = agg + (size_t)rowi * strideIn;
    const float* hr = hp  + (size_t)rowi * strideIn;
    float* orow = out + (size_t)rowi * strideOut;
    for (int c = threadIdx.x; c < Cout; c += blockDim.x) {
        float v = (ar[c] + hr[c]) * dv + b[c];
        orow[c] = fmaxf(v, 0.0f);
    }
}

// ---------------------------------------------------------------------------
extern "C" void kernel_launch(void* const* d_in, const int* in_sizes, int n_in,
                              void* d_out, int out_size, void* d_ws, size_t ws_size,
                              hipStream_t stream) {
    const float*     x  = (const float*)d_in[0];
    const long long* ei = (const long long*)d_in[1];
    const float*     W1 = (const float*)d_in[2];
    const float*     b1 = (const float*)d_in[3];
    const float*     W2 = (const float*)d_in[4];
    const float*     b2 = (const float*)d_in[5];
    float* out = (float*)d_out;

    const int N = N_NODES, E = N_EDGES;
    const int F1 = 512, F2P = 256, F2 = 250;

    // ---- workspace carve-up (256B aligned regions) ----
    char* ws = (char*)d_ws;
    size_t off = 0;
    auto take = [&](size_t bytes) -> void* {
        void* p = ws + off;
        off += (bytes + 255) & ~(size_t)255;
        return p;
    };
    float*     deg  = (float*)take((size_t)N * 4);
    float*     dinv = (float*)take((size_t)N * 4);
    _Float16*  W1sw = (_Float16*)take((size_t)128 * 512 * 2);
    _Float16*  W2sw = (_Float16*)take((size_t)512 * 256 * 2);
    float*     h1p  = (float*)take((size_t)N * F1 * 4);   // dinv-prescaled x@W1
    float*     agg1 = (float*)take((size_t)N * F1 * 4);   // layer-1 agg, then z1
    float*     h2p  = h1p;                                // reuse: h1p dead after post1
    float*     agg2 = (float*)((char*)h1p + (size_t)N * F2P * 4); // upper half of h1p
    (void)ws_size; (void)in_sizes; (void)n_in; (void)out_size;

    // ---- normalization ----
    hipMemsetAsync(deg, 0, (size_t)N * 4, stream);
    hipMemsetAsync(agg1, 0, (size_t)N * F1 * 4, stream);
    deg_count<<<(E + 255) / 256, 256, 0, stream>>>(ei, deg, E);
    dinv_kernel<<<(N + 255) / 256, 256, 0, stream>>>(deg, dinv, N);

    // ---- weight pre-swizzle to WMMA B-fragment layout (f16) ----
    swizzle_w<<<(4 * 32 * 512 + 255) / 256, 256, 0, stream>>>(W1, W1sw, 4, 32, 512);
    swizzle_w<<<(16 * 16 * 512 + 255) / 256, 256, 0, stream>>>(W2, W2sw, 16, 16, 250);

    // ---- layer 1 ----
    gcn_wmma_gemm<<<dim3(N / 16, 8), 32, 0, stream>>>(x, W1sw, dinv, h1p, 128, 32, F1);
    scatter_add<<<E, F1 / 4, 0, stream>>>(ei, h1p, agg1, F1, E);
    post_kernel<<<N, 256, 0, stream>>>(agg1, h1p, dinv, b1, agg1, F1, F1, F1); // z1 in-place

    // ---- layer 2 (agg2/h2p reuse h1p space; safe after post1) ----
    hipMemsetAsync(agg2, 0, (size_t)N * F2P * 4, stream);
    gcn_wmma_gemm<<<dim3(N / 16, 4), 32, 0, stream>>>(agg1, W2sw, dinv, h2p, 512, 16, F2P);
    scatter_add<<<E, F2P / 4, 0, stream>>>(ei, h2p, agg2, F2P, E);
    post_kernel<<<N, 256, 0, stream>>>(agg2, h2p, dinv, b2, out, F2P, F2, F2);
}
